// MoeBlock_53480932770319
// MI455X (gfx1250) — matrix operand order repaired and checked by reading, not compiled, not run on hardware
//
#include <hip/hip_runtime.h>
#include <math.h>

// ---- problem dims ----
#define T_TOK  4096
#define DIMD   768
#define HEADS  12
#define HDIM   64
#define NSEQ   1024
#define NBATCH 4
#define HIDDEN 3072
#define NEXP   8

// B-tile staging: 64 rows x 64 bf16 (128B) + 16B pad per row -> conflict-free, 16B-aligned
#define BSTRIDE 72
#define BBUF    (64 * BSTRIDE)

typedef __attribute__((ext_vector_type(16))) __bf16 v16bf;
typedef __attribute__((ext_vector_type(8)))  float  v8f;
typedef int v4i_ __attribute__((vector_size(16)));

union V16 { v16bf v; uint4 q[2]; };

static __device__ __forceinline__ v8f wmma_bf(const V16& a, const V16& b, v8f c) {
  return __builtin_amdgcn_wmma_f32_16x16x32_bf16(false, a.v, false, b.v,
                                                 (short)0, c, false, false);
}

// ---- CDNA5 async global->LDS copy (ASYNCcnt-tracked) ----
static __device__ __forceinline__ void async_copy16(const void* g, void* l) {
#if __has_builtin(__builtin_amdgcn_global_load_async_to_lds_b128)
  __builtin_amdgcn_global_load_async_to_lds_b128(
      (__attribute__((address_space(1))) v4i_*)(g),
      (__attribute__((address_space(3))) v4i_*)(l), 0, 0);
#else
  asm volatile("global_load_async_to_lds_b128 %0, %1, off"
               :: "v"((unsigned)(uintptr_t)(__attribute__((address_space(3))) void*)(l)),
                  "v"(g) : "memory");
#endif
}
static __device__ __forceinline__ void wait_async0() {
#if __has_builtin(__builtin_amdgcn_s_wait_asynccnt)
  __builtin_amdgcn_s_wait_asynccnt(0);
#else
  asm volatile("s_wait_asynccnt 0x0" ::: "memory");
#endif
}

// ---------------- f32 -> bf16 conversion ----------------
__global__ void cvt_kernel(const float* __restrict__ src, __bf16* __restrict__ dst, int n) {
  int i = blockIdx.x * blockDim.x + threadIdx.x;
  if (i < n) dst[i] = (__bf16)src[i];
}

// ---------------- LayerNorm: one wave per token ----------------
__global__ void ln_kernel(const float* __restrict__ x, const float* __restrict__ g,
                          const float* __restrict__ b, __bf16* __restrict__ out) {
  int wave = (blockIdx.x * blockDim.x + threadIdx.x) >> 5;
  int lane = threadIdx.x & 31;
  const float* row = x + (size_t)wave * DIMD;
  float vals[DIMD / 32];
  float s = 0.f, s2 = 0.f;
  #pragma unroll
  for (int i = 0; i < DIMD / 32; ++i) {
    float v = row[lane + 32 * i];
    vals[i] = v; s += v; s2 += v * v;
  }
  #pragma unroll
  for (int off = 16; off; off >>= 1) {
    s  += __shfl_xor(s,  off, 32);
    s2 += __shfl_xor(s2, off, 32);
  }
  float mean = s * (1.f / DIMD);
  float var  = s2 * (1.f / DIMD) - mean * mean;
  float rstd = rsqrtf(var + 1e-5f);
  __bf16* orow = out + (size_t)wave * DIMD;
  #pragma unroll
  for (int i = 0; i < DIMD / 32; ++i) {
    int c = lane + 32 * i;
    orow[c] = (__bf16)((vals[i] - mean) * rstd * g[c] + b[c]);
  }
}

// ---------------- block-cooperative pipelined bf16 GEMM core ----------------
// Block = 256 threads (8 waves). Computes C[128x64] tile:
//   wave w owns rows mblk+16w..+15; B tile (64n x 64k) staged in LDS, double-buffered
//   via async global->LDS copies (2x b128 per thread per stage); A operands register
//   double-buffered. 8 WMMAs per barrier; B operands batch-loaded to let the
//   scheduler overlap ds_loads with WMMA issue.
static __device__ __forceinline__ void gemm_block(const __bf16* __restrict__ A, int lda,
                                                  const __bf16* __restrict__ W, int ldw,
                                                  int mblk, int nbase, int K,
                                                  __bf16* bstage, v8f acc[4]) {
  int tid  = threadIdx.x;
  int wib  = tid >> 5, lane = tid & 31;
  int hl   = lane >> 4, l16 = lane & 15;
  int brow = tid >> 2, bchk = (tid & 3) * 8;   // 64 rows x 4x16B chunks (x2 k-halves)
  const __bf16* arow = A + (size_t)(mblk + wib * 16 + l16) * lda;
  const __bf16* wrow = W + (size_t)(nbase + brow) * ldw + bchk;
  __bf16* lrow = bstage + brow * BSTRIDE + bchk;

  const int nsteps = K / 64;
  // prologue: stage 64-K tile 0 into buf 0, preload A(0) chunks
  async_copy16(wrow, lrow);
  async_copy16(wrow + 32, lrow + 32);
  V16 a_cur[2], a_nxt[2];
  #pragma unroll
  for (int kc = 0; kc < 2; ++kc) {
    a_cur[kc].q[0] = *(const uint4*)(arow + 32 * kc + 8 * hl);
    a_cur[kc].q[1] = *(const uint4*)(arow + 32 * kc + 16 + 8 * hl);
  }

  for (int i = 0; i < nsteps; ++i) {
    int buf = i & 1;
    wait_async0();        // our slices of buf are in LDS
    __syncthreads();      // everyone's slices in LDS; prior reads of buf^1 done
    if (i + 1 < nsteps) {
      int kk = (i + 1) * 64;
      __bf16* lnx = bstage + (buf ^ 1) * BBUF + brow * BSTRIDE + bchk;
      async_copy16(wrow + kk, lnx);
      async_copy16(wrow + kk + 32, lnx + 32);
      #pragma unroll
      for (int kc = 0; kc < 2; ++kc) {
        a_nxt[kc].q[0] = *(const uint4*)(arow + kk + 32 * kc + 8 * hl);
        a_nxt[kc].q[1] = *(const uint4*)(arow + kk + 32 * kc + 16 + 8 * hl);
      }
    }
    const __bf16* bbuf = bstage + buf * BBUF + 8 * hl;
    #pragma unroll
    for (int kc = 0; kc < 2; ++kc) {
      V16 b[4];
      #pragma unroll
      for (int s = 0; s < 4; ++s) {
        const __bf16* bl = bbuf + (size_t)(16 * s + l16) * BSTRIDE + 32 * kc;
        b[s].q[0] = *(const uint4*)(bl);
        b[s].q[1] = *(const uint4*)(bl + 16);
      }
      #pragma unroll
      for (int s = 0; s < 4; ++s) acc[s] = wmma_bf(a_cur[kc], b[s], acc[s]);
    }
    a_cur[0] = a_nxt[0];
    a_cur[1] = a_nxt[1];
  }
}

// ---------------- QKV GEMM: qkv = h @ qkv_w^T, also Vt scatter ----------------
__global__ void qkv_kernel(const __bf16* __restrict__ h, const __bf16* __restrict__ wq,
                           __bf16* __restrict__ qkv, __bf16* __restrict__ vT) {
  __shared__ __align__(16) __bf16 bstage[2 * BBUF];
  const int NT = 2304 / 64;
  int mblk = (blockIdx.x / NT) * 128, nbase = (blockIdx.x % NT) * 64;
  int lane = threadIdx.x & 31, wib = threadIdx.x >> 5;
  v8f z = {};
  v8f acc[4] = {z, z, z, z};
  gemm_block(h, DIMD, wq, DIMD, mblk, nbase, DIMD, bstage, acc);
  int hl = lane >> 4, l16 = lane & 15;
  int mbase = mblk + wib * 16;
  #pragma unroll
  for (int s = 0; s < 4; ++s) {
    int n = nbase + 16 * s + l16;
    #pragma unroll
    for (int r = 0; r < 8; ++r) {
      int m = mbase + r + 8 * hl;
      float v = acc[s][r];
      qkv[(size_t)m * 2304 + n] = (__bf16)v;
      if (n >= 1536) {                    // V also stored transposed per (b,h)
        int hh = (n - 1536) >> 6, dd = (n - 1536) & 63;
        int bq = m >> 10, sp = m & 1023;
        vT[((size_t)(bq * HEADS + hh) * HDIM + dd) * NSEQ + sp] = (__bf16)v;
      }
    }
  }
}

// ---------------- flash attention: one wave per (b,h, 16-row q tile) ----------------
__global__ void attn_kernel(const __bf16* __restrict__ qkv, const __bf16* __restrict__ vT,
                            __bf16* __restrict__ aout) {
  __shared__ __align__(16) __bf16 ldsP[8][16 * 32];
  int wib  = threadIdx.x >> 5;
  int lane = threadIdx.x & 31;
  int gw   = blockIdx.x * (blockDim.x >> 5) + wib;
  const int QT = NSEQ / 16;
  int bh = gw / QT, qt = gw % QT;
  int b = bh / HEADS, hh = bh % HEADS;
  int hl = lane >> 4, l16 = lane & 15;

  const __bf16* qrow = qkv + (size_t)(b * NSEQ + qt * 16 + l16) * 2304 + hh * HDIM;
  V16 aq[2];
  #pragma unroll
  for (int c = 0; c < 2; ++c) {
    aq[c].q[0] = *(const uint4*)(qrow + 32 * c + 8 * hl);
    aq[c].q[1] = *(const uint4*)(qrow + 32 * c + 16 + 8 * hl);
  }

  v8f z = {};
  v8f o[4] = {z, z, z, z};
  float mrun[8], lrun[8];
  #pragma unroll
  for (int r = 0; r < 8; ++r) { mrun[r] = -1e30f; lrun[r] = 0.f; }
  const float scale = 0.125f;   // 1/sqrt(64)
  __bf16* myP = ldsP[wib];

  for (int step = 0; step < NSEQ / 32; ++step) {
    int kb = step * 32;
    v8f s[2];
    #pragma unroll
    for (int sub = 0; sub < 2; ++sub) {
      const __bf16* krow = qkv + (size_t)(b * NSEQ + kb + 16 * sub + l16) * 2304
                         + 768 + hh * HDIM + 8 * hl;
      V16 bk0, bk1;
      bk0.q[0] = *(const uint4*)(krow);
      bk0.q[1] = *(const uint4*)(krow + 16);
      bk1.q[0] = *(const uint4*)(krow + 32);
      bk1.q[1] = *(const uint4*)(krow + 48);
      v8f c = z;
      c = wmma_bf(aq[0], bk0, c);
      c = wmma_bf(aq[1], bk1, c);
      s[sub] = c;
    }
    float pr0[8], pr1[8];
    #pragma unroll
    for (int r = 0; r < 8; ++r) {
      float v0 = s[0][r] * scale, v1 = s[1][r] * scale;
      float mx = fmaxf(v0, v1);
      #pragma unroll
      for (int off = 8; off; off >>= 1) mx = fmaxf(mx, __shfl_xor(mx, off, 32));
      float mnew = fmaxf(mrun[r], mx);
      float corr = __expf(mrun[r] - mnew);
      float p0 = __expf(v0 - mnew), p1 = __expf(v1 - mnew);
      float rs = p0 + p1;
      #pragma unroll
      for (int off = 8; off; off >>= 1) rs += __shfl_xor(rs, off, 32);
      lrun[r] = lrun[r] * corr + rs;
      mrun[r] = mnew;
      pr0[r] = p0; pr1[r] = p1;
      #pragma unroll
      for (int j = 0; j < 4; ++j) o[j][r] *= corr;
    }
    #pragma unroll
    for (int r = 0; r < 8; ++r) {
      int m = r + 8 * hl;
      myP[m * 32 + l16]      = (__bf16)pr0[r];
      myP[m * 32 + 16 + l16] = (__bf16)pr1[r];
    }
    V16 pa;
    pa.q[0] = *(const uint4*)(myP + l16 * 32 + 8 * hl);
    pa.q[1] = *(const uint4*)(myP + l16 * 32 + 16 + 8 * hl);
    #pragma unroll
    for (int ns = 0; ns < 4; ++ns) {
      const __bf16* vrow = vT + ((size_t)bh * HDIM + 16 * ns + l16) * NSEQ + kb + 8 * hl;
      V16 bv;
      bv.q[0] = *(const uint4*)(vrow);
      bv.q[1] = *(const uint4*)(vrow + 16);
      o[ns] = wmma_bf(pa, bv, o[ns]);
    }
  }
  #pragma unroll
  for (int r = 0; r < 8; ++r) {
    float inv = 1.f / lrun[r];
    int m = qt * 16 + r + 8 * hl;
    #pragma unroll
    for (int ns = 0; ns < 4; ++ns) {
      int col = hh * HDIM + 16 * ns + l16;
      aout[(size_t)(b * NSEQ + m) * DIMD + col] = (__bf16)(o[ns][r] * inv);
    }
  }
}

// ---------------- proj GEMM + bias + residual -> f32 out ----------------
__global__ void proj_kernel(const __bf16* __restrict__ a, const __bf16* __restrict__ wp,
                            const float* __restrict__ pb, const float* __restrict__ x,
                            float* __restrict__ out) {
  __shared__ __align__(16) __bf16 bstage[2 * BBUF];
  const int NT = DIMD / 64;
  int mblk = (blockIdx.x / NT) * 128, nbase = (blockIdx.x % NT) * 64;
  int lane = threadIdx.x & 31, wib = threadIdx.x >> 5;
  v8f z = {};
  v8f acc[4] = {z, z, z, z};
  gemm_block(a, DIMD, wp, DIMD, mblk, nbase, DIMD, bstage, acc);
  int hl = lane >> 4, l16 = lane & 15;
  int mbase = mblk + wib * 16;
  #pragma unroll
  for (int s = 0; s < 4; ++s) {
    int n = nbase + 16 * s + l16;
    float bias = pb[n];
    #pragma unroll
    for (int r = 0; r < 8; ++r) {
      size_t idx = (size_t)(mbase + r + 8 * hl) * DIMD + n;
      out[idx] = x[idx] + acc[s][r] + bias;
    }
  }
}

// ---------------- gating: one wave per token, top-2 softmax ----------------
__global__ void gate_kernel(const __bf16* __restrict__ h2, const float* __restrict__ gw,
                            const float* __restrict__ gb, float* __restrict__ cw) {
  int t    = (blockIdx.x * blockDim.x + threadIdx.x) >> 5;
  int lane = threadIdx.x & 31;
  const __bf16* row = h2 + (size_t)t * DIMD;
  float logit[NEXP];
  #pragma unroll
  for (int e = 0; e < NEXP; ++e) {
    const float* w = gw + e * DIMD;
    float p = 0.f;
    for (int d = lane; d < DIMD; d += 32) p += (float)row[d] * w[d];
    #pragma unroll
    for (int off = 16; off; off >>= 1) p += __shfl_xor(p, off, 32);
    logit[e] = p + gb[e];
  }
  int i0 = 0;
  #pragma unroll
  for (int e = 1; e < NEXP; ++e) if (logit[e] > logit[i0]) i0 = e;
  int i1 = (i0 == 0) ? 1 : 0;
  #pragma unroll
  for (int e = 0; e < NEXP; ++e) if (e != i0 && logit[e] > logit[i1]) i1 = e;
  float e1 = __expf(logit[i1] - logit[i0]);
  float s0 = 1.f / (1.f + e1), s1 = e1 / (1.f + e1);
  if (lane < NEXP)
    cw[(size_t)t * NEXP + lane] = (lane == i0) ? s0 : ((lane == i1) ? s1 : 0.f);
}

// ---------------- FFN1: he = gelu(h2 @ w1[e]^T + b1[e]) ----------------
__global__ void ffn1_kernel(const __bf16* __restrict__ h2, const __bf16* __restrict__ w1e,
                            const float* __restrict__ b1e, __bf16* __restrict__ he) {
  __shared__ __align__(16) __bf16 bstage[2 * BBUF];
  const int NT = HIDDEN / 64;
  int mblk = (blockIdx.x / NT) * 128, nbase = (blockIdx.x % NT) * 64;
  int lane = threadIdx.x & 31, wib = threadIdx.x >> 5;
  v8f z = {};
  v8f acc[4] = {z, z, z, z};
  gemm_block(h2, DIMD, w1e, DIMD, mblk, nbase, DIMD, bstage, acc);
  int hl = lane >> 4, l16 = lane & 15;
  int mbase = mblk + wib * 16;
  #pragma unroll
  for (int s = 0; s < 4; ++s) {
    int n = nbase + 16 * s + l16;
    float bias = b1e[n];
    #pragma unroll
    for (int r = 0; r < 8; ++r) {
      float v = acc[s][r] + bias;
      v = 0.5f * v * (1.f + erff(v * 0.70710678118f));   // exact GELU
      he[(size_t)(mbase + r + 8 * hl) * HIDDEN + n] = (__bf16)v;
    }
  }
}

// ---------------- FFN2: out += cw[:,e] * (he @ w2[e]^T + b2[e]) ----------------
__global__ void ffn2_kernel(const __bf16* __restrict__ he, const __bf16* __restrict__ w2e,
                            const float* __restrict__ b2e, const float* __restrict__ cw,
                            int e, float* __restrict__ out) {
  __shared__ __align__(16) __bf16 bstage[2 * BBUF];
  const int NT = DIMD / 64;
  int mblk = (blockIdx.x / NT) * 128, nbase = (blockIdx.x % NT) * 64;
  int lane = threadIdx.x & 31, wib = threadIdx.x >> 5;
  v8f z = {};
  v8f acc[4] = {z, z, z, z};
  gemm_block(he, HIDDEN, w2e, HIDDEN, mblk, nbase, HIDDEN, bstage, acc);
  int hl = lane >> 4, l16 = lane & 15;
  int mbase = mblk + wib * 16;
  float w[8];
  #pragma unroll
  for (int r = 0; r < 8; ++r) w[r] = cw[(size_t)(mbase + r + 8 * hl) * NEXP + e];
  #pragma unroll
  for (int s = 0; s < 4; ++s) {
    int n = nbase + 16 * s + l16;
    float bias = b2e[n];
    #pragma unroll
    for (int r = 0; r < 8; ++r) {
      size_t idx = (size_t)(mbase + r + 8 * hl) * DIMD + n;
      out[idx] += w[r] * (acc[s][r] + bias);
    }
  }
}

// ---------------- host-side launch ----------------
extern "C" void kernel_launch(void* const* d_in, const int* in_sizes, int n_in,
                              void* d_out, int out_size, void* d_ws, size_t ws_size,
                              hipStream_t stream) {
  (void)in_sizes; (void)n_in; (void)out_size; (void)ws_size;
  const float* x      = (const float*)d_in[0];
  const float* ln1_g  = (const float*)d_in[1];
  const float* ln1_b  = (const float*)d_in[2];
  const float* qkv_w  = (const float*)d_in[3];
  const float* proj_w = (const float*)d_in[4];
  const float* proj_b = (const float*)d_in[5];
  const float* ln2_g  = (const float*)d_in[6];
  const float* ln2_b  = (const float*)d_in[7];
  const float* gate_w = (const float*)d_in[8];
  const float* gate_b = (const float*)d_in[9];
  const float* w1     = (const float*)d_in[10];
  const float* b1     = (const float*)d_in[11];
  const float* w2     = (const float*)d_in[12];
  const float* b2     = (const float*)d_in[13];
  float* out = (float*)d_out;

  char* ws = (char*)d_ws;
  auto carve = [&](size_t bytes) -> char* {
    char* p = ws; ws += (bytes + 255) & ~(size_t)255; return p;
  };
  __bf16* tok  = (__bf16*)carve((size_t)T_TOK * DIMD * 2);          // h, then h2
  __bf16* qkvb = (__bf16*)carve((size_t)T_TOK * 2304 * 2);
  __bf16* vT   = (__bf16*)carve((size_t)NBATCH * HEADS * HDIM * NSEQ * 2);
  __bf16* abuf = (__bf16*)carve((size_t)T_TOK * DIMD * 2);
  float*  cw   = (float*) carve((size_t)T_TOK * NEXP * 4);
  __bf16* wqb  = (__bf16*)carve((size_t)2304 * DIMD * 2);
  __bf16* wpb  = (__bf16*)carve((size_t)DIMD * DIMD * 2);
  __bf16* w1e  = (__bf16*)carve((size_t)HIDDEN * DIMD * 2);
  __bf16* w2e  = (__bf16*)carve((size_t)DIMD * HIDDEN * 2);
  __bf16* heb  = (__bf16*)carve((size_t)T_TOK * HIDDEN * 2);

  // weight conversions (f32 -> bf16)
  cvt_kernel<<<(2304 * DIMD + 255) / 256, 256, 0, stream>>>(qkv_w, wqb, 2304 * DIMD);
  cvt_kernel<<<(DIMD * DIMD + 255) / 256, 256, 0, stream>>>(proj_w, wpb, DIMD * DIMD);

  // attention branch
  ln_kernel<<<T_TOK / 8, 256, 0, stream>>>(x, ln1_g, ln1_b, tok);
  qkv_kernel<<<(T_TOK / 128) * (2304 / 64), 256, 0, stream>>>(tok, wqb, qkvb, vT);
  attn_kernel<<<NBATCH * HEADS * (NSEQ / 16) / 8, 256, 0, stream>>>(qkvb, vT, abuf);
  proj_kernel<<<(T_TOK / 128) * (DIMD / 64), 256, 0, stream>>>(abuf, wpb, proj_b, x, out);

  // MoE branch
  ln_kernel<<<T_TOK / 8, 256, 0, stream>>>(out, ln2_g, ln2_b, tok);
  gate_kernel<<<T_TOK / 8, 256, 0, stream>>>(tok, gate_w, gate_b, cw);
  for (int e = 0; e < NEXP; ++e) {
    cvt_kernel<<<(HIDDEN * DIMD + 255) / 256, 256, 0, stream>>>(
        w1 + (size_t)e * HIDDEN * DIMD, w1e, HIDDEN * DIMD);
    cvt_kernel<<<(DIMD * HIDDEN + 255) / 256, 256, 0, stream>>>(
        w2 + (size_t)e * DIMD * HIDDEN, w2e, DIMD * HIDDEN);
    ffn1_kernel<<<(T_TOK / 128) * (HIDDEN / 64), 256, 0, stream>>>(
        tok, w1e, b1 + (size_t)e * HIDDEN, heb);
    ffn2_kernel<<<(T_TOK / 128) * (DIMD / 64), 256, 0, stream>>>(
        heb, w2e, b2 + (size_t)e * DIMD, cw, e, out);
  }
}
